// point_wise_fn_accerator_4758823764027
// MI455X (gfx1250) — matrix operand order repaired and checked by reading, not compiled, not run on hardware
//
#include <hip/hip_runtime.h>

typedef __attribute__((ext_vector_type(8)))  __bf16 bf16x8;
typedef __attribute__((ext_vector_type(16))) __bf16 bf16x16;
typedef __attribute__((ext_vector_type(8)))  float  f32x8;
typedef __attribute__((ext_vector_type(4)))  float  f32x4;
typedef unsigned int u32;
typedef __attribute__((ext_vector_type(2))) u32 u32x2;
typedef __attribute__((ext_vector_type(4))) u32 u32x4;

static __device__ __forceinline__ unsigned short f2bf_rne(float f) {
  u32 u = __float_as_uint(f);
  u32 r = u + 0x7FFFu + ((u >> 16) & 1u);   // round-to-nearest-even
  return (unsigned short)(r >> 16);
}

// ---------------------------------------------------------------------------
// Kernel 1: collapse the tensor-train into dense W[n][k] (bf16), n in [0,1024),
// k in [0,512).  n = p*128 + q*16 + o ; k = i*64 + j*8 + kk
// W = sum_{r,s} a[p,i,r] * b[q,j,r,s] * c[o,kk,s]
// ---------------------------------------------------------------------------
__global__ __launch_bounds__(256) void build_w_kernel(
    const float* __restrict__ a, const float* __restrict__ b,
    const float* __restrict__ c, unsigned short* __restrict__ w) {
  int t  = blockIdx.x * 256 + threadIdx.x;   // 0 .. 524287
  int n  = t >> 9;
  int kk = t & 511;
  int p = n >> 7, q = (n >> 4) & 7, o = n & 15;
  int i = kk >> 6, j = (kk >> 3) & 7, k = kk & 7;
  float acc = 0.f;
#pragma unroll
  for (int r = 0; r < 2; ++r)
#pragma unroll
    for (int s = 0; s < 2; ++s)
      acc += a[(p * 8 + i) * 2 + r] * b[((q * 8 + j) * 2 + r) * 2 + s] *
             c[(o * 8 + k) * 2 + s];
  w[t] = f2bf_rne(acc);
}

// ---------------------------------------------------------------------------
// Kernel 2: Out[t, n] = relu( X[t, :] . W[n, :] + bias[n] )
// M = 65536, N = 1024, K = 512.  bf16 WMMA, f32 accumulate.
// Block: 256 threads = 8 waves laid out 4(M) x 2(N).
// Block M-tile = 128 rows; N processed in 8 chunks of 128 cols.
// Per-wave tile: 32(M) x 64(N) -> 8 accumulators; each K-step: 2 A-frags +
// 4 B-frags -> 8 WMMAs, B reused against both A frags in registers.
// ---------------------------------------------------------------------------
#define XROWS 128
#define WCOLS 128
#define LSTR  520   // 512 + 8 pad elems -> 1040B row stride (bank-conflict free)

__global__ __launch_bounds__(256) void tt_gemm_kernel(
    const float* __restrict__ x, const unsigned short* __restrict__ w,
    const float* __restrict__ bias, float* __restrict__ out) {
  __shared__ unsigned short lx[XROWS * LSTR];  // 133,120 B
  __shared__ unsigned short lw[WCOLS * LSTR];  // 133,120 B   (total 266,240 B)

  const int tid  = threadIdx.x;
  const int wave = tid >> 5;
  const int lane = tid & 31;
  const int hi   = lane >> 4;        // selects K-half per ISA 16-bit layouts
  const int l15  = lane & 15;
  const int wm   = wave & 3;         // M-wave: rows [32*wm, 32*wm+32)
  const int wn   = wave >> 2;        // N-wave: cols [64*wn, 64*wn+64) in chunk
  const long m0  = (long)blockIdx.x * XROWS;

  // ---- stage X tile: 128x512 f32 -> bf16 in LDS (read-once: nontemporal) ----
  {
    const float* xb = x + m0 * 512;
#pragma unroll 1
    for (int idx = tid * 4; idx < XROWS * 512; idx += 256 * 4) {
      int row = idx >> 9, col = idx & 511;
      f32x4 v = __builtin_nontemporal_load((const f32x4*)(xb + row * 512 + col));
      u32x2 pk;
      pk[0] = (u32)f2bf_rne(v[0]) | ((u32)f2bf_rne(v[1]) << 16);
      pk[1] = (u32)f2bf_rne(v[2]) | ((u32)f2bf_rne(v[3]) << 16);
      *(u32x2*)(&lx[row * LSTR + col]) = pk;
    }
  }

  // A-fragment row bases for this lane (ISA 16-bit A 16x32 layout: M = lane&15)
  const unsigned short* axp0 = &lx[(wm * 32 + l15) * LSTR];
  const unsigned short* axp1 = &lx[(wm * 32 + 16 + l15) * LSTR];
  const int akb = hi * 8;    // A K-subbase
  const int bkb = hi * 8;    // B K-subbase

  for (int nc = 0; nc < 8; ++nc) {
    // ---- stage W chunk: 128x512 bf16 (coalesced 16B copies) ----
    const unsigned short* wsrc = w + nc * (WCOLS * 512);
#pragma unroll 1
    for (int idx = tid * 8; idx < WCOLS * 512; idx += 256 * 8) {
      int row = idx >> 9, col = idx & 511;
      *(u32x4*)(&lw[row * LSTR + col]) = *(const u32x4*)(wsrc + idx);
    }
    __syncthreads();

    // ---- prefetch next W chunk into cache while we compute this one ----
    {
      const unsigned short* wnext = w + ((nc + 1) & 7) * (WCOLS * 512);
#pragma unroll
      for (int pf = 0; pf < 4; ++pf)
        __builtin_prefetch(wnext + tid * 64 + pf * 16384, 0, 1);
    }

    f32x8 acc[8];
#pragma unroll
    for (int t = 0; t < 8; ++t)
      acc[t] = (f32x8){0.f, 0.f, 0.f, 0.f, 0.f, 0.f, 0.f, 0.f};

#pragma unroll 2
    for (int kc = 0; kc < 16; ++kc) {
      const int kof = kc * 32;
      bf16x8 a0lo = *(const bf16x8*)(axp0 + kof + akb);
      bf16x8 a0hi = *(const bf16x8*)(axp0 + kof + akb + 16);
      bf16x8 a1lo = *(const bf16x8*)(axp1 + kof + akb);
      bf16x8 a1hi = *(const bf16x8*)(axp1 + kof + akb + 16);
      bf16x16 A0 = __builtin_shufflevector(a0lo, a0hi, 0, 1, 2, 3, 4, 5, 6, 7,
                                           8, 9, 10, 11, 12, 13, 14, 15);
      bf16x16 A1 = __builtin_shufflevector(a1lo, a1hi, 0, 1, 2, 3, 4, 5, 6, 7,
                                           8, 9, 10, 11, 12, 13, 14, 15);
#pragma unroll
      for (int t = 0; t < 4; ++t) {
        const unsigned short* bp =
            &lw[(wn * 64 + t * 16 + l15) * LSTR + kof + bkb];
        bf16x8 blo = *(const bf16x8*)(bp);
        bf16x8 bhi = *(const bf16x8*)(bp + 16);
        bf16x16 B = __builtin_shufflevector(blo, bhi, 0, 1, 2, 3, 4, 5, 6, 7,
                                            8, 9, 10, 11, 12, 13, 14, 15);
        acc[t]     = __builtin_amdgcn_wmma_f32_16x16x32_bf16(
            false, A0, false, B, (short)0, acc[t], false, false);
        acc[4 + t] = __builtin_amdgcn_wmma_f32_16x16x32_bf16(
            false, A1, false, B, (short)0, acc[4 + t], false, false);
      }
    }

    // ---- epilogue: bias + relu + nontemporal store ----
#pragma unroll
    for (int s = 0; s < 2; ++s) {
      const int mb = wm * 32 + s * 16 + hi * 8;  // D layout: VGPR v -> row mb+v
#pragma unroll
      for (int t = 0; t < 4; ++t) {
        const int ng = nc * 128 + wn * 64 + t * 16 + l15;  // D: col = lane&15
        const float bv = bias[ng];
        float* orow = out + (m0 + mb) * 1024 + ng;
#pragma unroll
        for (int v = 0; v < 8; ++v) {
          float o = fmaxf(acc[s * 4 + t][v] + bv, 0.f);
          __builtin_nontemporal_store(o, orow + (long)v * 1024);
        }
      }
    }
    __syncthreads();  // protect lw before next chunk overwrites it
  }
}

// ---------------------------------------------------------------------------
extern "C" void kernel_launch(void* const* d_in, const int* in_sizes, int n_in,
                              void* d_out, int out_size, void* d_ws,
                              size_t ws_size, hipStream_t stream) {
  const float* x    = (const float*)d_in[0];  // [16,4096,512]
  const float* a    = (const float*)d_in[1];  // [8,8,2]
  const float* b    = (const float*)d_in[2];  // [8,8,2,2]
  const float* c    = (const float*)d_in[3];  // [16,8,2]
  const float* bias = (const float*)d_in[4];  // [8,8,16] -> 1024
  unsigned short* w = (unsigned short*)d_ws;  // 1024*512 bf16 = 1 MB scratch

  build_w_kernel<<<2048, 256, 0, stream>>>(a, b, c, w);               // 524288
  tt_gemm_kernel<<<512, 256, 0, stream>>>(x, w, bias, (float*)d_out); // 65536/128
}
